// StageBlock_36361193128264
// MI455X (gfx1250) — matrix operand order repaired and checked by reading, not compiled, not run on hardware
//
#include <hip/hip_runtime.h>

// ---------------------------------------------------------------------------
// MS-Deformable-Attention stage block for MI455X (gfx1250, wave32, WMMA).
// GEMMs use v_wmma_f32_16x16x32_bf16 (bf16 in, f32 accumulate) with 2x4
// register tiling (32x64 C tile / wave, 8 WMMAs per K-step).
// ---------------------------------------------------------------------------

typedef __attribute__((ext_vector_type(16))) __bf16 v16bf;
typedef __attribute__((ext_vector_type(8)))  float  v8f;

#define NH   12
#define NL   4
#define NP   4
#define DH   64
#define DM   768
#define BATCH 4
#define LQ   1600
#define LV   5440
#define MQ   (BATCH * LQ)   // 6400
#define MV   (BATCH * LV)   // 21760

__device__ __forceinline__ unsigned short f2bf(float f) {
    unsigned u = __float_as_uint(f);
    unsigned r = u + 0x7FFFu + ((u >> 16) & 1u);   // round-to-nearest-even
    return (unsigned short)(r >> 16);
}

union FragU { uint4 u[2]; v16bf v; };

// A fragment: 16x32 bf16 tile, per ISA table:
// lanes 0-15  -> row M = lane,    K in {0..7, 16..23}
// lanes 16-31 -> row M = lane-16, K in {8..15, 24..31}
__device__ __forceinline__ v16bf load_a_frag(const unsigned short* __restrict__ base,
                                             int K, int row, int k, int lh) {
    const unsigned short* p = base + (size_t)row * K + k + lh * 8;
    FragU f;
    f.u[0] = *reinterpret_cast<const uint4*>(p);        // K = klo .. klo+7
    f.u[1] = *reinterpret_cast<const uint4*>(p + 16);   // K = klo+16 .. klo+23
    return f.v;
}

// B fragment: 32x16 bf16 tile, read from W^T [N,K] (row N = output column):
// lanes 0-15 -> N = lane, K {0..7,16..23} ; lanes 16-31 -> N = lane-16, K {8..15,24..31}
__device__ __forceinline__ v16bf load_b_frag(const unsigned short* __restrict__ base,
                                             int K, int col, int k, int lh) {
    const unsigned short* p = base + (size_t)col * K + k + lh * 8;
    FragU f;
    f.u[0] = *reinterpret_cast<const uint4*>(p);
    f.u[1] = *reinterpret_cast<const uint4*>(p + 16);
    return f.v;
}

// ---------------------------------------------------------------------------
// Weight transpose + f32->bf16:  Wt[n*K+k] = bf16(W[k*N+n])
// ---------------------------------------------------------------------------
__global__ __launch_bounds__(256) void transpose_bf16_kernel(
    const float* __restrict__ W, unsigned short* __restrict__ Wt, int K, int N) {
    int idx = blockIdx.x * blockDim.x + threadIdx.x;
    if (idx >= K * N) return;
    int n = idx / K, k = idx - n * K;
    Wt[idx] = f2bf(W[(size_t)k * N + n]);
}

// ---------------------------------------------------------------------------
// LayerNorm over D=768, one block (256 threads) per row, bf16 output.
// ---------------------------------------------------------------------------
__global__ __launch_bounds__(256) void layernorm_bf16_kernel(
    const float* __restrict__ in, const float* __restrict__ scale,
    const float* __restrict__ bias, unsigned short* __restrict__ out) {
    __shared__ float s1[256], s2[256];
    int row = blockIdx.x;
    int tid = threadIdx.x;
    const float* p = in + (size_t)row * DM;
    float x0 = p[tid], x1 = p[tid + 256], x2 = p[tid + 512];
    s1[tid] = x0 + x1 + x2;
    s2[tid] = x0 * x0 + x1 * x1 + x2 * x2;
    __syncthreads();
    for (int o = 128; o > 0; o >>= 1) {
        if (tid < o) { s1[tid] += s1[tid + o]; s2[tid] += s2[tid + o]; }
        __syncthreads();
    }
    float mean = s1[0] * (1.0f / DM);
    float var  = s2[0] * (1.0f / DM) - mean * mean;
    float rstd = rsqrtf(var + 1e-6f);
    unsigned short* q = out + (size_t)row * DM;
    q[tid]       = f2bf((x0 - mean) * rstd * scale[tid]       + bias[tid]);
    q[tid + 256] = f2bf((x1 - mean) * rstd * scale[tid + 256] + bias[tid + 256]);
    q[tid + 512] = f2bf((x2 - mean) * rstd * scale[tid + 512] + bias[tid + 512]);
}

// ---------------------------------------------------------------------------
// WMMA GEMM: C = A[M,K](bf16) @ Wt[N,K]^T(bf16) + bias, f32 accumulate.
// One 32x64 C tile per wave (2 M-subtiles x 4 N-subtiles); 8 WMMAs / K-step.
// MODE 0: plain f32 store C[row*N+n]
// MODE 1: value projection, scatter to value[B][nH][Lv][dh]
// MODE 2: output projection epilogue: prompt = (acc+b)*gamma,
//         out0 = x + prompt, out1 = prompt
// Requires: M % 32 == 0, N % 64 == 0, K % 32 == 0 (and LV % 32 == 0 for MODE 1).
// ---------------------------------------------------------------------------
template <int MODE>
__global__ __launch_bounds__(256) void wmma_gemm_bf16_kernel(
    const unsigned short* __restrict__ A, const unsigned short* __restrict__ Bt,
    const float* __restrict__ bias, float* __restrict__ C,
    int M, int N, int K, int tilesN,
    const float* __restrict__ x, const float* __restrict__ gamma,
    float* __restrict__ C2) {
    int wave = blockIdx.x * (blockDim.x >> 5) + (threadIdx.x >> 5);
    int lane = threadIdx.x & 31;
    int mt = wave / tilesN;
    int nt = wave - mt * tilesN;
    if (mt * 32 >= M) return;
    int lh = lane >> 4;     // half-wave select
    int ll = lane & 15;

    v8f c[2][4];
    #pragma unroll
    for (int s = 0; s < 2; ++s)
        #pragma unroll
        for (int j = 0; j < 4; ++j)
            #pragma unroll
            for (int i = 0; i < 8; ++i) c[s][j][i] = 0.0f;

    int arow0 = mt * 32 + ll;           // M-subtile 0 row for this lane
    int nbase = nt * 64;
    int bcol  = nbase + ll;             // N-subtile j col = bcol + j*16

    for (int k = 0; k < K; k += 32) {
        // Prefetch the A stream a few K-steps ahead (global_prefetch_b8).
        if (k + 128 < K) {
            __builtin_prefetch(A + (size_t)arow0 * K + k + 128, 0, 3);
            __builtin_prefetch(A + (size_t)(arow0 + 16) * K + k + 128, 0, 3);
        }
        v16bf a[2], b[4];
        a[0] = load_a_frag(A, K, arow0,      k, lh);
        a[1] = load_a_frag(A, K, arow0 + 16, k, lh);
        #pragma unroll
        for (int j = 0; j < 4; ++j) b[j] = load_b_frag(Bt, K, bcol + j * 16, k, lh);
        #pragma unroll
        for (int s = 0; s < 2; ++s)
            #pragma unroll
            for (int j = 0; j < 4; ++j)
                c[s][j] = __builtin_amdgcn_wmma_f32_16x16x32_bf16(
                    false, a[s], false, b[j], (short)0, c[s][j], false, false);
    }

    // C/D layout: lane ll -> column n; lanes 0-15 rows 0..7, lanes 16-31 rows 8..15
    int b_idx = 0, lv0 = 0;
    if (MODE == 1) { b_idx = (mt * 32) / LV; lv0 = mt * 32 - b_idx * LV; }
    #pragma unroll
    for (int j = 0; j < 4; ++j) {
        int n  = nbase + j * 16 + ll;
        float bn = bias[n];
        #pragma unroll
        for (int s = 0; s < 2; ++s) {
            int mb = mt * 32 + s * 16 + lh * 8;
            #pragma unroll
            for (int i = 0; i < 8; ++i) {
                int row = mb + i;
                float v = c[s][j][i] + bn;
                if (MODE == 0) {
                    C[(size_t)row * N + n] = v;
                } else if (MODE == 1) {
                    int lv = lv0 + s * 16 + lh * 8 + i;
                    int h = n >> 6, d = n & 63;
                    C[(((size_t)(b_idx * NH + h)) * LV + lv) * DH + d] = v;
                } else {
                    float p = v * gamma[n];
                    size_t idx = (size_t)row * N + n;
                    C[idx]  = x[idx] + p;   // x_out
                    C2[idx] = p;            // prompt
                }
            }
        }
    }
}

// ---------------------------------------------------------------------------
// Softmax over 16 attention logits per (b,q,h), in place.
// ---------------------------------------------------------------------------
__global__ __launch_bounds__(256) void softmax16_kernel(float* __restrict__ a, int total) {
    int t = blockIdx.x * blockDim.x + threadIdx.x;
    if (t >= total) return;
    float* p = a + (size_t)t * 16;
    float m = p[0];
    #pragma unroll
    for (int i = 1; i < 16; ++i) m = fmaxf(m, p[i]);
    float s = 0.0f;
    #pragma unroll
    for (int i = 0; i < 16; ++i) { float e = __expf(p[i] - m); p[i] = e; s += e; }
    float inv = 1.0f / s;
    #pragma unroll
    for (int i = 0; i < 16; ++i) p[i] *= inv;
}

// ---------------------------------------------------------------------------
// Deformable bilinear sampling: one wave per (b,q,h); each lane owns 2 of the
// 64 head channels (float2 = contiguous, fully coalesced per corner row).
// Writes bf16 directly (A matrix of the final GEMM).
// ---------------------------------------------------------------------------
__device__ __forceinline__ void corner_acc(const float* __restrict__ vl,
                                           int Wl, int Hl, int xi, int yi,
                                           float ww, int d0, float& a0, float& a1) {
    float s = ((xi >= 0) & (xi < Wl) & (yi >= 0) & (yi < Hl)) ? 1.0f : 0.0f;
    int xc = xi < 0 ? 0 : (xi > Wl - 1 ? Wl - 1 : xi);
    int yc = yi < 0 ? 0 : (yi > Hl - 1 ? Hl - 1 : yi);
    const float2 v = *reinterpret_cast<const float2*>(vl + ((size_t)(yc * Wl + xc)) * DH + d0);
    float t = ww * s;
    a0 += t * v.x;
    a1 += t * v.y;
}

__global__ __launch_bounds__(256) void deform_sample_kernel(
    const float* __restrict__ value,    // [B,nH,Lv,dh]
    const float* __restrict__ refpts,   // [B,Lq,nL,2]
    const float* __restrict__ offsets,  // [B,Lq,nH,nL,nP,2]
    const float* __restrict__ attw,     // [B,Lq,nH,nL*nP]
    unsigned short* __restrict__ outbf) // [B,Lq,D] bf16
{
    const int HL[NL] = {64, 32, 16, 8};
    const int WL[NL] = {64, 32, 16, 8};
    const int ST[NL] = {0, 4096, 5120, 5376};

    int wave = blockIdx.x * (blockDim.x >> 5) + (threadIdx.x >> 5);
    int lane = threadIdx.x & 31;
    if (wave >= BATCH * LQ * NH) return;
    int h  = wave % NH;
    int bq = wave / NH;                 // b*Lq + q
    int b  = bq / LQ;

    const float* vbase = value + ((size_t)(b * NH + h)) * LV * DH;
    const float* off   = offsets + (((size_t)bq * NH + h) * NL * NP) * 2;
    const float* aw    = attw + ((size_t)bq * NH + h) * (NL * NP);
    const float* rp    = refpts + (size_t)bq * NL * 2;

    float acc0 = 0.0f, acc1 = 0.0f;
    int d0 = lane * 2;

    #pragma unroll
    for (int l = 0; l < NL; ++l) {
        int Hl = HL[l], Wl = WL[l];
        float fW = (float)Wl, fH = (float)Hl;
        const float* vl = vbase + (size_t)ST[l] * DH;
        float rx = rp[l * 2 + 0], ry = rp[l * 2 + 1];
        #pragma unroll
        for (int p = 0; p < NP; ++p) {
            int pi = l * NP + p;
            float ox = off[pi * 2 + 0], oy = off[pi * 2 + 1];
            float px = (rx + ox / fW) * fW - 0.5f;
            float py = (ry + oy / fH) * fH - 0.5f;
            float w  = aw[pi];
            float x0f = floorf(px), y0f = floorf(py);
            float wx = px - x0f, wy = py - y0f;
            int x0 = (int)x0f, y0 = (int)y0f;
            corner_acc(vl, Wl, Hl, x0,     y0,     w * (1.0f - wx) * (1.0f - wy), d0, acc0, acc1);
            corner_acc(vl, Wl, Hl, x0 + 1, y0,     w * wx          * (1.0f - wy), d0, acc0, acc1);
            corner_acc(vl, Wl, Hl, x0,     y0 + 1, w * (1.0f - wx) * wy,          d0, acc0, acc1);
            corner_acc(vl, Wl, Hl, x0 + 1, y0 + 1, w * wx          * wy,          d0, acc0, acc1);
        }
    }

    unsigned pack = (unsigned)f2bf(acc0) | ((unsigned)f2bf(acc1) << 16);
    *reinterpret_cast<unsigned*>(outbf + (size_t)bq * DM + h * DH + d0) = pack;
}

// ---------------------------------------------------------------------------
// Host launch
// ---------------------------------------------------------------------------
static inline int cdiv(int a, int b) { return (a + b - 1) / b; }

extern "C" void kernel_launch(void* const* d_in, const int* in_sizes, int n_in,
                              void* d_out, int out_size, void* d_ws, size_t ws_size,
                              hipStream_t stream) {
    const float* x        = (const float*)d_in[0];
    const float* feat     = (const float*)d_in[1];
    const float* refpts   = (const float*)d_in[2];
    // d_in[3] spatial_shapes, d_in[4] level_start_index, d_in[5] H, d_in[6] W: static per setup
    const float* qn_scale = (const float*)d_in[7];
    const float* qn_bias  = (const float*)d_in[8];
    const float* fn_scale = (const float*)d_in[9];
    const float* fn_bias  = (const float*)d_in[10];
    const float* gamma    = (const float*)d_in[11];
    const float* Wv       = (const float*)d_in[12];
    const float* bv       = (const float*)d_in[13];
    const float* Wo       = (const float*)d_in[14];
    const float* bo       = (const float*)d_in[15];
    const float* Wa       = (const float*)d_in[16];
    const float* ba       = (const float*)d_in[17];
    const float* Wout     = (const float*)d_in[18];
    const float* bout     = (const float*)d_in[19];

    char* ws = (char*)d_ws;
    size_t o = 0;
    unsigned short* q_bf   = (unsigned short*)(ws + o); o += (size_t)MQ * DM * 2;
    unsigned short* f_bf   = (unsigned short*)(ws + o); o += (size_t)MV * DM * 2;
    float*          value  = (float*)         (ws + o); o += (size_t)MV * DM * 4;
    unsigned short* Wv_t   = (unsigned short*)(ws + o); o += (size_t)DM * DM * 2;
    unsigned short* Wo_t   = (unsigned short*)(ws + o); o += (size_t)DM * (NH*NL*NP*2) * 2;
    unsigned short* Wa_t   = (unsigned short*)(ws + o); o += (size_t)DM * (NH*NL*NP) * 2;
    unsigned short* Wout_t = (unsigned short*)(ws + o); o += (size_t)DM * DM * 2;
    float*          offs   = (float*)         (ws + o); o += (size_t)MQ * (NH*NL*NP*2) * 4;
    float*          attw   = (float*)         (ws + o); o += (size_t)MQ * (NH*NL*NP) * 4;
    unsigned short* out_bf = (unsigned short*)(ws + o); o += (size_t)MQ * DM * 2;

    // 1) weight transpose + bf16
    transpose_bf16_kernel<<<cdiv(DM * DM, 256), 256, 0, stream>>>(Wv, Wv_t, DM, DM);
    transpose_bf16_kernel<<<cdiv(DM * 384, 256), 256, 0, stream>>>(Wo, Wo_t, DM, 384);
    transpose_bf16_kernel<<<cdiv(DM * 192, 256), 256, 0, stream>>>(Wa, Wa_t, DM, 192);
    transpose_bf16_kernel<<<cdiv(DM * DM, 256), 256, 0, stream>>>(Wout, Wout_t, DM, DM);

    // 2) layernorms -> bf16
    layernorm_bf16_kernel<<<MQ, 256, 0, stream>>>(x,    qn_scale, qn_bias, q_bf);
    layernorm_bf16_kernel<<<MV, 256, 0, stream>>>(feat, fn_scale, fn_bias, f_bf);

    // 3) value projection (scatter to [B,nH,Lv,dh])
    {
        int tilesN = DM / 64, tiles = (MV / 32) * tilesN;
        wmma_gemm_bf16_kernel<1><<<cdiv(tiles, 8), 256, 0, stream>>>(
            f_bf, Wv_t, bv, value, MV, DM, DM, tilesN, nullptr, nullptr, nullptr);
    }
    // 4) offsets GEMM
    {
        int N = NH * NL * NP * 2, tilesN = N / 64, tiles = (MQ / 32) * tilesN;
        wmma_gemm_bf16_kernel<0><<<cdiv(tiles, 8), 256, 0, stream>>>(
            q_bf, Wo_t, bo, offs, MQ, N, DM, tilesN, nullptr, nullptr, nullptr);
    }
    // 5) attention logits GEMM + softmax over 16
    {
        int N = NH * NL * NP, tilesN = N / 64, tiles = (MQ / 32) * tilesN;
        wmma_gemm_bf16_kernel<0><<<cdiv(tiles, 8), 256, 0, stream>>>(
            q_bf, Wa_t, ba, attw, MQ, N, DM, tilesN, nullptr, nullptr, nullptr);
        softmax16_kernel<<<cdiv(MQ * NH, 256), 256, 0, stream>>>(attw, MQ * NH);
    }
    // 6) deformable bilinear sampling -> bf16 out
    deform_sample_kernel<<<cdiv(MQ * NH, 8), 256, 0, stream>>>(
        value, refpts, offs, attw, out_bf);

    // 7) output projection with fused gamma/residual epilogue -> (x_out, prompt)
    {
        float* x_out  = (float*)d_out;
        float* prompt = (float*)d_out + (size_t)MQ * DM;
        int tilesN = DM / 64, tiles = (MQ / 32) * tilesN;
        wmma_gemm_bf16_kernel<2><<<cdiv(tiles, 8), 256, 0, stream>>>(
            out_bf, Wout_t, bout, x_out, MQ, DM, DM, tilesN, x, gamma, prompt);
    }
}
